// Block_12799002542237
// MI455X (gfx1250) — compile-verified, matrix operand order checked by reading, not run on hardware
//
#include <hip/hip_runtime.h>
#include <math.h>

// ---------------------------------------------------------------------------
// MI455X (gfx1250) implementation of the Mamba/MMSE block.
// All dense channel-mixing GEMMs go through v_wmma_f32_16x16x32_f16 (wave32,
// 16x64 output strip per wave: A fragment loaded once per K-step, reused by
// 4 WMMAs; f32 accumulate). The jmax==4 fast path is branch-free inside the
// K loop so no exec-mask churn surrounds the WMMAs. Elementwise / reduction
// stages are plain VALU kernels (bandwidth bound on 23.3 TB/s HBM).
// ---------------------------------------------------------------------------

typedef __attribute__((ext_vector_type(16))) _Float16 v16h;
typedef __attribute__((ext_vector_type(8)))  _Float16 v8h;
typedef __attribute__((ext_vector_type(8)))  float    v8f;

#define EPSF 1e-5f

__device__ __forceinline__ v16h ld16h(const _Float16* p) {
  v8h a0 = *(const v8h*)(p);
  v8h a1 = *(const v8h*)(p + 16);
  return __builtin_shufflevector(a0, a1, 0,1,2,3,4,5,6,7,8,9,10,11,12,13,14,15);
}

// ---------------------------------------------------------------------------
// Generic WMMA GEMM:  C[M,N] = A[M,K] * W[N,K]^T  (A,W f16 row-major)
// Each wave computes a 16x64 strip (4 n-tiles; wave-uniform jmax handles the
// N=48 tail). epilogue: v = v*scale[n]*scaleMul + bias[n]; optional clip;
// optional +resid; outMode 0: row-major [M,N] f32; outMode 1: NCHW scatter.
// A layout per lane: row = tm + (lane&15), K-half = lane>>4 (8 halves at
// k0+8*kh and 8 at k0+16+8*kh) -> two 128-bit loads. B symmetric with N.
// ---------------------------------------------------------------------------
__global__ void k_gemm(const _Float16* __restrict__ A, const _Float16* __restrict__ W,
                       float* __restrict__ C, int M, int N, int K,
                       const float* __restrict__ scale, float scaleMul,
                       const float* __restrict__ bias,
                       const float* __restrict__ resid,
                       int doClip, float clipLo, float clipHi, int outMode) {
  const int lane = threadIdx.x & 31;
  const int wv   = threadIdx.x >> 5;
  const int ntN  = N >> 4;                       // 16-wide n-tiles
  const int ngN  = (ntN + 3) >> 2;               // groups of 4 n-tiles
  const int tile = blockIdx.x * (blockDim.x >> 5) + wv;
  if (tile >= (M >> 4) * ngN) return;            // wave-uniform
  const int tm   = (tile / ngN) << 4;
  const int gq   = tile % ngN;
  const int tn0  = gq << 6;
  const int rem  = ntN - (gq << 2);
  const int jmax = rem < 4 ? rem : 4;            // wave-uniform
  const int kh   = lane >> 4;
  const int ml   = lane & 15;

  const _Float16* pa = A + (size_t)(tm + ml) * K + 8 * kh;
  const _Float16* pb = W + (size_t)(tn0 + ml) * K + 8 * kh;
  const size_t bstr = (size_t)16 * K;

  const v8f z8 = {0.f, 0.f, 0.f, 0.f, 0.f, 0.f, 0.f, 0.f};
  v8f acc0 = z8, acc1 = z8, acc2 = z8, acc3 = z8;

  if (jmax == 4) {
    // ---- branch-free fast path (all GEMMs except the N=48 tail) ----
    for (int k0 = 0; k0 < K; k0 += 32) {
      v16h av = ld16h(pa + k0);
      if (k0 + 32 < K) {                         // gfx1250 global_prefetch_b8
        __builtin_prefetch(pa + k0 + 32, 0, 1);
        __builtin_prefetch(pb + k0 + 32, 0, 1);
      }
      v16h bv0 = ld16h(pb + k0);
      v16h bv1 = ld16h(pb + bstr + k0);
      v16h bv2 = ld16h(pb + 2 * bstr + k0);
      v16h bv3 = ld16h(pb + 3 * bstr + k0);
      acc0 = __builtin_amdgcn_wmma_f32_16x16x32_f16(false, av, false, bv0,
                                                    (short)0, acc0, false, false);
      acc1 = __builtin_amdgcn_wmma_f32_16x16x32_f16(false, av, false, bv1,
                                                    (short)0, acc1, false, false);
      acc2 = __builtin_amdgcn_wmma_f32_16x16x32_f16(false, av, false, bv2,
                                                    (short)0, acc2, false, false);
      acc3 = __builtin_amdgcn_wmma_f32_16x16x32_f16(false, av, false, bv3,
                                                    (short)0, acc3, false, false);
    }
  } else {
    // ---- guarded tail path (x_proj: N=48 -> jmax==3) ----
    for (int k0 = 0; k0 < K; k0 += 32) {
      v16h av = ld16h(pa + k0);
      {
        v16h bv = ld16h(pb + k0);
        acc0 = __builtin_amdgcn_wmma_f32_16x16x32_f16(false, av, false, bv,
                                                      (short)0, acc0, false, false);
      }
      if (jmax > 1) {
        v16h bv = ld16h(pb + bstr + k0);
        acc1 = __builtin_amdgcn_wmma_f32_16x16x32_f16(false, av, false, bv,
                                                      (short)0, acc1, false, false);
      }
      if (jmax > 2) {
        v16h bv = ld16h(pb + 2 * bstr + k0);
        acc2 = __builtin_amdgcn_wmma_f32_16x16x32_f16(false, av, false, bv,
                                                      (short)0, acc2, false, false);
      }
    }
  }

  auto epi = [&](v8f a, int ncol) {
    const float sc = scale ? scale[ncol] * scaleMul : 1.0f;
    const float bs = bias ? bias[ncol] : 0.0f;
#pragma unroll
    for (int i = 0; i < 8; ++i) {
      const int mrow = tm + i + 8 * kh;          // C layout: vgpr i -> M=i(+8)
      float v = a[i] * sc + bs;
      if (doClip) v = fminf(fmaxf(v, clipLo), clipHi);
      if (resid) v += resid[(size_t)mrow * N + ncol];
      if (outMode == 0) {
        C[(size_t)mrow * N + ncol] = v;
      } else {                                   // NCHW scatter (H=W=64)
        const int b  = mrow >> 12;
        const int hw = mrow & 4095;
        C[(((size_t)b * N + ncol) << 12) + hw] = v;
      }
    }
  };
  epi(acc0, tn0 + ml);
  if (jmax > 1) epi(acc1, tn0 + 16 + ml);
  if (jmax > 2) epi(acc2, tn0 + 32 + ml);
  if (jmax > 3) epi(acc3, tn0 + 48 + ml);
}

// fp32 -> f16 weight staging
__global__ void k_f2h(const float* __restrict__ s, _Float16* __restrict__ d, int n) {
  int i = blockIdx.x * 256 + threadIdx.x;
  if (i < n) d[i] = (_Float16)s[i];
}

// ---------------------------------------------------------------------------
// upsample depthwise: stride-2 transposed 3x3 conv (lhs_dilation=2, pad(1,2),
// flipped weights) + BN + ReLU -> f16 NHWC stage for up_pw GEMM.
// ---------------------------------------------------------------------------
__global__ void k_updw(const float* __restrict__ x, const float* __restrict__ w,
                       const float* __restrict__ g, const float* __restrict__ bb,
                       _Float16* __restrict__ out) {
  int t = blockIdx.x * 256 + threadIdx.x;        // 8*128*128*256
  int c = t & 255;
  int m = t >> 8;                                // b*16384 + oy*128 + ox
  int ox = m & 127, oy = (m >> 7) & 127, b = m >> 14;
  float acc = 0.f;
  for (int ky = 0; ky < 3; ++ky) {
    int jy = oy + ky - 1;
    if (jy < 0 || jy > 126 || (jy & 1)) continue;
    int iy = jy >> 1;
    for (int kx = 0; kx < 3; ++kx) {
      int jx = ox + kx - 1;
      if (jx < 0 || jx > 126 || (jx & 1)) continue;
      int ix = jx >> 1;
      acc += w[c * 9 + (2 - ky) * 3 + (2 - kx)] *
             x[(((size_t)b * 256 + c) * 64 + iy) * 64 + ix];
    }
  }
  float rs = rsqrtf(1.0f + EPSF);
  float v = fmaxf(acc * g[c] * rs + bb[c], 0.f);
  out[(size_t)m * 256 + c] = (_Float16)v;
}

// ---------------------------------------------------------------------------
// row/col max & min (+argmax/argmin) branches, + interp1d(pos_embed) + idx.
// Writes raw (pre-LN) sequences: [b, l(=0..63 col | 64..127 row), c].
// ---------------------------------------------------------------------------
__global__ void k_branch(const float* __restrict__ x, const float* __restrict__ pe,
                         float* __restrict__ smax, float* __restrict__ smin) {
  int t = blockIdx.x * 256 + threadIdx.x;        // 8*64*256
  int c = t & 255;
  int j = (t >> 8) & 63;
  int b = t >> 14;
  const float* xb = x + ((size_t)b * 256 + c) * 4096;
  float cmx = -1e30f, cmn = 1e30f; int cai = 0, cii = 0;
  for (int h = 0; h < 64; ++h) {
    float v = xb[h * 64 + j];
    if (v > cmx) { cmx = v; cai = h; }
    if (v < cmn) { cmn = v; cii = h; }
  }
  float rmx = -1e30f, rmn = 1e30f; int rai = 0, rii = 0;
  for (int w = 0; w < 64; ++w) {
    float v = xb[j * 64 + w];
    if (v > rmx) { rmx = v; rai = w; }
    if (v < rmn) { rmn = v; rii = w; }
  }
  float coord = fminf(fmaxf(j * 0.25f - 0.375f, 0.f), 15.f);
  int lo = (int)floorf(coord);
  int hi = lo + 1 > 15 ? 15 : lo + 1;
  float wq = coord - (float)lo;
  float p = pe[c * 16 + lo] * (1.f - wq) + pe[c * 16 + hi] * wq;
  smax[((size_t)b * 128 + j) * 256 + c]      = cmx + p + (float)cai;
  smax[((size_t)b * 128 + 64 + j) * 256 + c] = rmx + p + (float)rai;
  smin[((size_t)b * 128 + j) * 256 + c]      = cmn + p + (float)cii;
  smin[((size_t)b * 128 + 64 + j) * 256 + c] = rmn + p + (float)rii;
}

// wave-level LayerNorm over 256 channels (8 values/lane)
__device__ __forceinline__ void ln256_wave(const float* __restrict__ src,
                                           const float* __restrict__ g,
                                           const float* __restrict__ be,
                                           _Float16* d16a, _Float16* d16b,
                                           float* df) {
  int lane = threadIdx.x & 31;
  float v[8]; float s = 0.f;
#pragma unroll
  for (int i = 0; i < 8; ++i) { v[i] = src[lane * 8 + i]; s += v[i]; }
  for (int off = 16; off; off >>= 1) s += __shfl_xor(s, off, 32);
  float mean = s * (1.0f / 256.0f);
  float vs = 0.f;
#pragma unroll
  for (int i = 0; i < 8; ++i) { float d = v[i] - mean; vs += d * d; }
  for (int off = 16; off; off >>= 1) vs += __shfl_xor(vs, off, 32);
  float inv = rsqrtf(vs * (1.0f / 256.0f) + EPSF);
#pragma unroll
  for (int i = 0; i < 8; ++i) {
    int c = lane * 8 + i;
    float o = (v[i] - mean) * inv * g[c] + be[c];
    if (d16a) d16a[c] = (_Float16)o;
    if (d16b) d16b[c] = (_Float16)o;
    if (df)   df[c]   = o;
  }
}

// LN each seq row -> forward f16 stage + per-batch-reversed f16 stage
__global__ void k_ln_stage(const float* __restrict__ src, const float* __restrict__ g,
                           const float* __restrict__ be,
                           _Float16* __restrict__ fwd, _Float16* __restrict__ rev) {
  int row = blockIdx.x * (blockDim.x >> 5) + (threadIdx.x >> 5); // 1024 rows
  int b = row >> 7, l = row & 127;
  ln256_wave(src + (size_t)row * 256, g, be,
             fwd + (size_t)row * 256,
             rev + ((size_t)(b * 128 + 127 - l)) * 256, nullptr);
}

// causal depthwise conv1d (k=4, left pad 3) + bias + SiLU
__global__ void k_conv1d(const float* __restrict__ xz, const float* __restrict__ cw,
                         const float* __restrict__ cb, float* __restrict__ xsf,
                         _Float16* __restrict__ xsh) {
  int t = blockIdx.x * 256 + threadIdx.x;        // 1024*512
  int d = t & 511, m = t >> 9;                   // m = b*128 + l
  int b = m >> 7, l = m & 127;
  float acc = cb[d];
  for (int tt = 0; tt < 4; ++tt) {
    int li = l + tt - 3;
    if (li < 0) continue;
    acc += cw[d * 4 + tt] * xz[((size_t)(b * 128 + li)) * 1024 + d];
  }
  float s = acc / (1.f + expf(-acc));
  xsf[(size_t)m * 512 + d] = s;
  xsh[(size_t)m * 512 + d] = (_Float16)s;
}

// dt = softplus(dbl[:, :16] @ dt_w^T + dt_b)   (K=16: plain VALU, tiny)
__global__ void k_dt(const float* __restrict__ dbl, const float* __restrict__ dw,
                     const float* __restrict__ db, float* __restrict__ dt) {
  int t = blockIdx.x * 256 + threadIdx.x;        // 1024*512
  int d = t & 511, m = t >> 9;
  float acc = db[d];
  const float* r = dbl + (size_t)m * 48;
#pragma unroll
  for (int k = 0; k < 16; ++k) acc += r[k] * dw[d * 16 + k];
  dt[(size_t)m * 512 + d] = acc > 20.f ? acc : log1pf(expf(acc));
}

// selective scan: 16 state-lanes per (b,d); shfl-reduce h*C over s
__global__ void k_scan(const float* __restrict__ dbl, const float* __restrict__ dt,
                       const float* __restrict__ xs, const float* __restrict__ Alog,
                       float* __restrict__ ys) {
  int t = blockIdx.x * 256 + threadIdx.x;        // 4096 pairs * 16 states
  int s = t & 15;
  int pair = t >> 4;                             // b*512 + d
  int d = pair & 511, b = pair >> 9;
  float Av = -expf(Alog[d * 16 + s]);
  float h = 0.f;
  for (int l = 0; l < 128; ++l) {
    size_t m = (size_t)b * 128 + l;
    float dtv = dt[m * 512 + d];
    float Bv  = dbl[m * 48 + 16 + s];
    float Cv  = dbl[m * 48 + 32 + s];
    float xv  = xs[m * 512 + d];
    h = expf(dtv * Av) * h + dtv * Bv * xv;
    float p = h * Cv;
    p += __shfl_xor(p, 1, 16);
    p += __shfl_xor(p, 2, 16);
    p += __shfl_xor(p, 4, 16);
    p += __shfl_xor(p, 8, 16);
    if (s == 0) ys[m * 512 + d] = p;
  }
}

// y = (ys + xs*D) * silu(zz) -> f16 stage for out_proj
__global__ void k_gate(const float* __restrict__ ys, const float* __restrict__ xs,
                       const float* __restrict__ xz, const float* __restrict__ Dv,
                       _Float16* __restrict__ yh) {
  int t = blockIdx.x * 256 + threadIdx.x;        // 1024*512
  int d = t & 511, m = t >> 9;
  float y = ys[(size_t)m * 512 + d] + xs[(size_t)m * 512 + d] * Dv[d];
  float z = xz[(size_t)m * 1024 + 512 + d];
  y *= z / (1.f + expf(-z));
  yh[(size_t)m * 512 + d] = (_Float16)y;
}

// cat = [ln(fwd_out), reverse(ln(bwd_out))] -> f16 [row,512]
__global__ void k_lncat(const float* __restrict__ mf, const float* __restrict__ mb,
                        const float* __restrict__ g, const float* __restrict__ be,
                        _Float16* __restrict__ cat) {
  int row = blockIdx.x * (blockDim.x >> 5) + (threadIdx.x >> 5); // 1024 rows
  int b = row >> 7, l = row & 127;
  ln256_wave(mf + (size_t)row * 256, g, be, cat + (size_t)row * 512, nullptr, nullptr);
  ln256_wave(mb + ((size_t)(b * 128 + 127 - l)) * 256, g, be,
             cat + (size_t)row * 512 + 256, nullptr, nullptr);
}

__global__ void k_lnrow(const float* __restrict__ src, const float* __restrict__ g,
                        const float* __restrict__ be, float* __restrict__ dst) {
  int row = blockIdx.x * (blockDim.x >> 5) + (threadIdx.x >> 5); // 1024 rows
  ln256_wave(src + (size_t)row * 256, g, be, nullptr, nullptr, dst + (size_t)row * 256);
}

// res = clip(rmax[b,y,c] + rmin[b,x,c] + 3, 0, 6)/6 * z   (in place on z, NHWC)
__global__ void k_scale_res(float* __restrict__ z, const float* __restrict__ rmx,
                            const float* __restrict__ rmn) {
  int t = blockIdx.x * 256 + threadIdx.x;        // 131072*256
  int c = t & 255, m = t >> 8;                   // b*16384 + y*128 + x
  int xp = m & 127, y = (m >> 7) & 127, b = m >> 14;
  float v = rmx[((size_t)b * 128 + y) * 256 + c] +
            rmn[((size_t)b * 128 + xp) * 256 + c] + 3.f;
  v = fminf(fmaxf(v, 0.f), 6.f) * (1.f / 6.f);
  z[(size_t)m * 256 + c] *= v;
}

// GroupNorm stats (32 groups of 8 ch), NHWC input, one block per (b,g)
__global__ void k_gn_stats(const float* __restrict__ src, int npix,
                           float* __restrict__ mean, float* __restrict__ inv) {
  __shared__ float sh[256], sh2[256];
  int bg = blockIdx.x;                           // b*32 + g
  int g = bg & 31, b = bg >> 5;
  float s = 0.f, s2 = 0.f;
  for (int pix = threadIdx.x; pix < npix; pix += 256) {
    const float* p = src + ((size_t)b * npix + pix) * 256 + g * 8;
#pragma unroll
    for (int ci = 0; ci < 8; ++ci) { float v = p[ci]; s += v; s2 += v * v; }
  }
  sh[threadIdx.x] = s; sh2[threadIdx.x] = s2;
  __syncthreads();
  for (int off = 128; off; off >>= 1) {
    if (threadIdx.x < off) { sh[threadIdx.x] += sh[threadIdx.x + off];
                             sh2[threadIdx.x] += sh2[threadIdx.x + off]; }
    __syncthreads();
  }
  if (threadIdx.x == 0) {
    float n = (float)npix * 8.f;
    float m0 = sh[0] / n;
    mean[bg] = m0;
    inv[bg]  = rsqrtf(sh2[0] / n - m0 * m0 + EPSF);
  }
}

// apply GN to 128x128 NHWC, sample even pixels -> f16 [b*4096+oy*64+ox, c]
__global__ void k_stage_down(const float* __restrict__ z, const float* __restrict__ mean,
                             const float* __restrict__ inv, const float* __restrict__ gg,
                             const float* __restrict__ gb, _Float16* __restrict__ out) {
  int t = blockIdx.x * 256 + threadIdx.x;        // 32768*256
  int c = t & 255, m = t >> 8;
  int ox = m & 63, oy = (m >> 6) & 63, b = m >> 12;
  int pix = (oy * 2) * 128 + ox * 2;
  int bg = b * 32 + (c >> 3);
  float v = (z[((size_t)b * 16384 + pix) * 256 + c] - mean[bg]) * inv[bg] * gg[c] + gb[c];
  out[(size_t)m * 256 + c] = (_Float16)v;
}

// out = GN(src) + x_nchw ; optional f16 copy
__global__ void k_gn_add(const float* __restrict__ src, const float* __restrict__ mean,
                         const float* __restrict__ inv, const float* __restrict__ gg,
                         const float* __restrict__ gb, const float* __restrict__ x,
                         float* __restrict__ out, _Float16* __restrict__ outh) {
  int t = blockIdx.x * 256 + threadIdx.x;        // 32768*256
  int c = t & 255, m = t >> 8;
  int b = m >> 12, hw = m & 4095;
  int bg = b * 32 + (c >> 3);
  float v = (src[(size_t)m * 256 + c] - mean[bg]) * inv[bg] * gg[c] + gb[c];
  v += x[(((size_t)b * 256 + c) << 12) + hw];
  out[(size_t)m * 256 + c] = v;
  if (outh) outh[(size_t)m * 256 + c] = (_Float16)v;
}

// mlp depthwise 3x3 (pad 1) + bias + ReLU on NHWC[...,512] -> f16 stage
__global__ void k_dwconv(const float* __restrict__ h1, const float* __restrict__ w,
                         const float* __restrict__ bb, _Float16* __restrict__ out) {
  int t = blockIdx.x * 256 + threadIdx.x;        // 32768*512
  int c = t & 511, m = t >> 9;
  int wx = m & 63, hy = (m >> 6) & 63, b = m >> 12;
  float acc = bb[c];
  for (int dy = -1; dy <= 1; ++dy) {
    int yy = hy + dy; if (yy < 0 || yy > 63) continue;
    for (int dx = -1; dx <= 1; ++dx) {
      int xx = wx + dx; if (xx < 0 || xx > 63) continue;
      acc += w[c * 9 + (dy + 1) * 3 + (dx + 1)] *
             h1[(((size_t)b * 4096) + yy * 64 + xx) * 512 + c];
    }
  }
  out[(size_t)m * 512 + c] = (_Float16)fmaxf(acc, 0.f);
}

// ---------------------------------------------------------------------------
// Host orchestration
// ---------------------------------------------------------------------------
struct MambaP {
  const float *in_proj, *conv_w, *conv_b, *x_proj, *dt_w, *dt_b, *A_log, *D, *out_proj;
};
struct MambaH { _Float16 *in_proj, *x_proj, *out_proj; };
struct MambaWS {
  float *xz, *xsf, *dbl, *dt, *ys;
  _Float16 *xsh, *ygate;
};

static void gemm(hipStream_t st, const _Float16* A, const _Float16* W, float* C,
                 int M, int N, int K, const float* scale, float scaleMul,
                 const float* bias, const float* resid, int doClip,
                 float lo, float hi, int outMode) {
  int tiles = (M >> 4) * (((N >> 4) + 3) >> 2);  // 16x64 strips per wave
  int blocks = (tiles + 7) / 8;
  k_gemm<<<blocks, 256, 0, st>>>(A, W, C, M, N, K, scale, scaleMul, bias,
                                 resid, doClip, lo, hi, outMode);
}

static void run_mamba(hipStream_t st, const _Float16* seq16, const MambaP& p,
                      const MambaH& ph, const MambaWS& w, float* mout) {
  gemm(st, seq16, ph.in_proj, w.xz, 1024, 1024, 256, nullptr, 1.f, nullptr,
       nullptr, 0, 0, 0, 0);
  k_conv1d<<<2048, 256, 0, st>>>(w.xz, p.conv_w, p.conv_b, w.xsf, w.xsh);
  gemm(st, w.xsh, ph.x_proj, w.dbl, 1024, 48, 512, nullptr, 1.f, nullptr,
       nullptr, 0, 0, 0, 0);
  k_dt<<<2048, 256, 0, st>>>(w.dbl, p.dt_w, p.dt_b, w.dt);
  k_scan<<<256, 256, 0, st>>>(w.dbl, w.dt, w.xsf, p.A_log, w.ys);
  k_gate<<<2048, 256, 0, st>>>(w.ys, w.xsf, w.xz, p.D, w.ygate);
  gemm(st, w.ygate, ph.out_proj, mout, 1024, 256, 512, nullptr, 1.f, nullptr,
       nullptr, 0, 0, 0, 0);
}

extern "C" void kernel_launch(void* const* d_in, const int* in_sizes, int n_in,
                              void* d_out, int out_size, void* d_ws, size_t ws_size,
                              hipStream_t stream) {
  (void)in_sizes; (void)n_in; (void)out_size; (void)ws_size;
  auto F = [&](int i) { return (const float*)d_in[i]; };

  const float* x      = F(0);
  const float* pe     = F(1);
  const float* ln_g   = F(2);
  const float* ln_b   = F(3);
  MambaP fwd = { F(4), F(5), F(6), F(7), F(8), F(9), F(10), F(11), F(12) };
  MambaP bwd = { F(13), F(14), F(15), F(16), F(17), F(18), F(19), F(20), F(21) };
  const float* out_w  = F(22);
  const float* out_b  = F(23);
  const float* up_dw  = F(24);
  const float* updw_g = F(25);
  const float* updw_b = F(26);
  const float* up_pw  = F(27);
  const float* uppw_g = F(28);
  const float* uppw_b = F(29);
  const float* dn_pw  = F(30);
  const float* dn_g   = F(31);
  const float* dn_b   = F(32);
  const float* gn_g   = F(33);
  const float* gn_b   = F(34);
  const float* fc1_w  = F(35);
  const float* fc1_g  = F(36);
  const float* fc1_b  = F(37);
  const float* dw_w   = F(38);
  const float* dw_b   = F(39);
  const float* fc2_w  = F(40);
  const float* fc2_g  = F(41);
  const float* fc2_b  = F(42);
  float* out = (float*)d_out;

  // ---- workspace carve (bump allocator, 256B aligned) ----
  char* base = (char*)d_ws;
  size_t off = 0;
  auto alloc = [&](size_t bytes) -> void* {
    void* p = base + off;
    off = (off + bytes + 255) & ~(size_t)255;
    return p;
  };
  _Float16* up_h   = (_Float16*)alloc(67108864);   // regionA: 131072x256 f16
  float*    zbuf   = (float*)   alloc(134217728);  // regionB: 131072x256 f32
  float*    sm_raw = (float*)   alloc(1048576);
  float*    sn_raw = (float*)   alloc(1048576);
  _Float16* smF    = (_Float16*)alloc(524288);
  _Float16* smR    = (_Float16*)alloc(524288);
  _Float16* snF    = (_Float16*)alloc(524288);
  _Float16* snR    = (_Float16*)alloc(524288);
  MambaWS mws;
  mws.xz    = (float*)   alloc(4194304);
  mws.xsf   = (float*)   alloc(2097152);
  mws.xsh   = (_Float16*)alloc(1048576);
  mws.dbl   = (float*)   alloc(196608);
  mws.dt    = (float*)   alloc(2097152);
  mws.ys    = (float*)   alloc(2097152);
  mws.ygate = (_Float16*)alloc(1048576);
  float* moutF  = (float*)alloc(1048576);
  float* moutB  = (float*)alloc(1048576);
  _Float16* cat16 = (_Float16*)alloc(1048576);
  float* bid    = (float*)alloc(1048576);
  float* rmaxln = (float*)alloc(1048576);
  float* rminln = (float*)alloc(1048576);
  float* mean1 = (float*)alloc(1024); float* inv1 = (float*)alloc(1024);
  float* mean2 = (float*)alloc(1024); float* inv2 = (float*)alloc(1024);
  float* mean3 = (float*)alloc(1024); float* inv3 = (float*)alloc(1024);
  _Float16* Adown = (_Float16*)alloc(16777216);    // 32768x256 f16
  float* d1  = (float*)alloc(33554432);            // 32768x256 f32
  float* x1f = (float*)alloc(33554432);            // 32768x256 f32
  // f16 weights
  _Float16* uppw_h = (_Float16*)alloc(131072);
  _Float16* inF_h  = (_Float16*)alloc(524288);
  _Float16* inB_h  = (_Float16*)alloc(524288);
  _Float16* xpF_h  = (_Float16*)alloc(49152);
  _Float16* xpB_h  = (_Float16*)alloc(49152);
  _Float16* opF_h  = (_Float16*)alloc(262144);
  _Float16* opB_h  = (_Float16*)alloc(262144);
  _Float16* outw_h = (_Float16*)alloc(262144);
  _Float16* dn_h   = (_Float16*)alloc(131072);
  _Float16* fc1_h  = (_Float16*)alloc(262144);
  _Float16* fc2_h  = (_Float16*)alloc(262144);
  // regionA reuse (after up_pw GEMM): d3 + x1h.  regionB reuse: h1 + h2.
  float*    d3  = (float*)up_h;
  _Float16* x1h = (_Float16*)((char*)up_h + 33554432);
  float*    h1  = zbuf;
  _Float16* h2  = (_Float16*)((char*)zbuf + 67108864);

  MambaH fwdH = { inF_h, xpF_h, opF_h };
  MambaH bwdH = { inB_h, xpB_h, opB_h };
  const float bnMul = 1.0f / sqrtf(1.0f + EPSF);

  // ---- weight staging to f16 ----
  auto f2h = [&](const float* s, _Float16* d, int n) {
    k_f2h<<<(n + 255) / 256, 256, 0, stream>>>(s, d, n);
  };
  f2h(up_pw, uppw_h, 65536);
  f2h(fwd.in_proj, inF_h, 262144);   f2h(bwd.in_proj, inB_h, 262144);
  f2h(fwd.x_proj, xpF_h, 24576);     f2h(bwd.x_proj, xpB_h, 24576);
  f2h(fwd.out_proj, opF_h, 131072);  f2h(bwd.out_proj, opB_h, 131072);
  f2h(out_w, outw_h, 131072);
  f2h(dn_pw, dn_h, 65536);
  f2h(fc1_w, fc1_h, 131072);
  f2h(fc2_w, fc2_h, 131072);

  // ---- upsample path: dw-transposed conv + BN + ReLU, then pw GEMM + BN + clip ----
  k_updw<<<131072, 256, 0, stream>>>(x, up_dw, updw_g, updw_b, up_h);
  gemm(stream, up_h, uppw_h, zbuf, 131072, 256, 256, uppw_g, bnMul, uppw_b,
       nullptr, 1, 0.f, 6.f, 0);                 // z = clip(bn(pw),0,6)  NHWC

  // ---- branch sequences + LN staging ----
  k_branch<<<512, 256, 0, stream>>>(x, pe, sm_raw, sn_raw);
  k_ln_stage<<<128, 256, 0, stream>>>(sm_raw, ln_g, ln_b, smF, smR);
  k_ln_stage<<<128, 256, 0, stream>>>(sn_raw, ln_g, ln_b, snF, snR);

  // ---- bidirectional mamba on seq_max -> rmaxln ----
  run_mamba(stream, smF, fwd, fwdH, mws, moutF);
  run_mamba(stream, smR, bwd, bwdH, mws, moutB);
  k_lncat<<<128, 256, 0, stream>>>(moutF, moutB, ln_g, ln_b, cat16);
  gemm(stream, cat16, outw_h, bid, 1024, 256, 512, nullptr, 1.f, out_b,
       nullptr, 0, 0, 0, 0);
  k_lnrow<<<128, 256, 0, stream>>>(bid, ln_g, ln_b, rmaxln);

  // ---- bidirectional mamba on seq_min -> rminln ----
  run_mamba(stream, snF, fwd, fwdH, mws, moutF);
  run_mamba(stream, snR, bwd, bwdH, mws, moutB);
  k_lncat<<<128, 256, 0, stream>>>(moutF, moutB, ln_g, ln_b, cat16);
  gemm(stream, cat16, outw_h, bid, 1024, 256, 512, nullptr, 1.f, out_b,
       nullptr, 0, 0, 0, 0);
  k_lnrow<<<128, 256, 0, stream>>>(bid, ln_g, ln_b, rminln);

  // ---- res = clip(rmax+rmin+3,0,6)/6 * z ; GN ; stride-2 pw ; BN ; GN ; +x ; GN ----
  k_scale_res<<<131072, 256, 0, stream>>>(zbuf, rmaxln, rminln);
  k_gn_stats<<<256, 256, 0, stream>>>(zbuf, 16384, mean1, inv1);
  k_stage_down<<<32768, 256, 0, stream>>>(zbuf, mean1, inv1, gn_g, gn_b, Adown);
  gemm(stream, Adown, dn_h, d1, 32768, 256, 256, dn_g, bnMul, dn_b,
       nullptr, 0, 0, 0, 0);                     // bn(conv stride2), NHWC
  k_gn_stats<<<256, 256, 0, stream>>>(d1, 4096, mean2, inv2);
  k_gn_add<<<32768, 256, 0, stream>>>(d1, mean2, inv2, gn_g, gn_b, x, d3, nullptr);
  k_gn_stats<<<256, 256, 0, stream>>>(d3, 4096, mean3, inv3);
  k_gn_add<<<32768, 256, 0, stream>>>(d3, mean3, inv3, gn_g, gn_b, x, x1f, x1h);
  // x1 = x + gn(gn(bn(conv)) + x)  == x + mmse(x)

  // ---- MLP: fc1 + BN, dw3x3 + ReLU, fc2 + BN, + x1, scatter NCHW ----
  gemm(stream, x1h, fc1_h, h1, 32768, 512, 256, fc1_g, bnMul, fc1_b,
       nullptr, 0, 0, 0, 0);
  k_dwconv<<<65536, 256, 0, stream>>>(h1, dw_w, dw_b, h2);
  gemm(stream, h2, fc2_h, out, 32768, 256, 512, fc2_g, bnMul, fc2_b,
       x1f, 0, 0, 0, 1);                         // out = x1 + bn(fc2), NCHW
}
